// DirectAU_73735998537818
// MI455X (gfx1250) — compile-verified, matrix operand order checked by reading, not run on hardware
//
#include <hip/hip_runtime.h>
#include <hip/hip_bf16.h>
#include <math.h>

typedef __attribute__((ext_vector_type(16))) _Float16 v16h;
typedef __attribute__((ext_vector_type(8)))  _Float16 v8h;
typedef __attribute__((ext_vector_type(4)))  _Float16 v4h;
typedef __attribute__((ext_vector_type(8)))  float    v8f;

#define N_ROWS 8192
#define DIM 64
#define NTILE 512                       // N_ROWS / 16 (16-row tiles)
#define TILES_PER_STRIP 4               // each wave: 16 rows x 64 cols
#define NSTRIPC (NTILE / TILES_PER_STRIP)      // 128 strip columns
#define NSTRIPS (NTILE * NSTRIPC)              // 65536 strips (full grid)
#define WAVES_PER_BLOCK 8
#define EPS_U 1e-8f
// exp(-2*max(2-2g,0)) = exp2(min((4g-4)*log2(e), 0));  C4 = 4*log2(e)
#define C4 5.770780163555852f

struct HalfPair { v8h lo, hi; };
union V16U { v16h v; HalfPair p; };

// ---------------------------------------------------------------------------
// Kernel 1: per-row fp32 normalization, fp16 emission for WMMA, and the
// alignment term mean ||u_hat - i_hat||^2 in fp32. One thread per row.
// ---------------------------------------------------------------------------
__global__ void norm_align_kernel(const float* __restrict__ U,
                                  const float* __restrict__ I,
                                  _Float16* __restrict__ U16,
                                  _Float16* __restrict__ I16,
                                  float* __restrict__ alignPart) {
  __shared__ float red[256];
  const int row = blockIdx.x * blockDim.x + threadIdx.x;
  const float* u = U + (size_t)row * DIM;
  const float* w = I + (size_t)row * DIM;

  float su = 0.f, si = 0.f;
#pragma unroll
  for (int d = 0; d < DIM; d += 4) {
    float4 a = *(const float4*)(u + d);
    float4 b = *(const float4*)(w + d);
    su += a.x * a.x + a.y * a.y + a.z * a.z + a.w * a.w;
    si += b.x * b.x + b.y * b.y + b.z * b.z + b.w * b.w;
  }
  const float rnu = 1.0f / sqrtf(su);
  const float rni = 1.0f / sqrtf(si);

  float d2 = 0.f;
#pragma unroll
  for (int d = 0; d < DIM; d += 4) {
    float4 a = *(const float4*)(u + d);
    float4 b = *(const float4*)(w + d);
    float ux = a.x * rnu, uy = a.y * rnu, uz = a.z * rnu, uw = a.w * rnu;
    float ix = b.x * rni, iy = b.y * rni, iz = b.z * rni, iw = b.w * rni;
    float dx = ux - ix, dy = uy - iy, dz = uz - iz, dw = uw - iw;
    d2 += dx * dx + dy * dy + dz * dz + dw * dw;
    v4h hu = { (_Float16)ux, (_Float16)uy, (_Float16)uz, (_Float16)uw };
    v4h hi = { (_Float16)ix, (_Float16)iy, (_Float16)iz, (_Float16)iw };
    *(v4h*)(U16 + (size_t)row * DIM + d) = hu;
    *(v4h*)(I16 + (size_t)row * DIM + d) = hi;
  }

  red[threadIdx.x] = d2;
  __syncthreads();
  for (int s = 128; s > 0; s >>= 1) {
    if (threadIdx.x < s) red[threadIdx.x] += red[threadIdx.x + s];
    __syncthreads();
  }
  if (threadIdx.x == 0) alignPart[blockIdx.x] = red[0];
}

// ---------------------------------------------------------------------------
// Operand fetch for v_wmma_f32_16x16x32_f16 (A layout; Gram B uses the same
// per-lane pattern since B = X^T with lane = output column = row of X).
// kc selects K-chunk [0,32) or [32,64).
// ---------------------------------------------------------------------------
__device__ __forceinline__ v16h load_operand(const _Float16* __restrict__ rowPtr,
                                             int hi, int kc) {
  V16U r;
  const _Float16* base = rowPtr + kc * 32 + hi * 8;
  r.p.lo = *(const v8h*)(base);        // K = kc*32 + hi*8 + [0..7]
  r.p.hi = *(const v8h*)(base + 16);   // K = kc*32 + 16 + hi*8 + [0..7]
  return r.v;
}

// ---------------------------------------------------------------------------
// Kernel 2: one wave per 16x64 strip of the Gram matrix (A reused across 4
// B tiles -> 8 WMMAs / 20 b128 loads). Upper-triangle weighting per sub-tile
// (0 below diag, 1 on diag, 2 above). Raw v_exp_f32 post-processing (argument
// range [-11.6, 0] needs no OCML denormal fixup), wave32 butterfly reduction,
// per-strip partial to workspace (deterministic). blockIdx.y selects matrix.
// ---------------------------------------------------------------------------
__global__ void gram_exp_kernel(const _Float16* __restrict__ Xu,
                                const _Float16* __restrict__ Xi,
                                float* __restrict__ partU,
                                float* __restrict__ partI) {
  const _Float16* __restrict__ X = (blockIdx.y == 0) ? Xu : Xi;
  float* __restrict__ part = (blockIdx.y == 0) ? partU : partI;

  const int wave = threadIdx.x >> 5;
  const int lane = threadIdx.x & 31;
  const unsigned sid = blockIdx.x * WAVES_PER_BLOCK + wave;
  const unsigned r  = sid >> 7;                  // / NSTRIPC
  const unsigned cs = sid & (NSTRIPC - 1);
  const unsigned c0 = cs * TILES_PER_STRIP;      // first 16-col tile index

  if (c0 + (TILES_PER_STRIP - 1) < r) {          // strip fully below diagonal
    if (lane == 0) part[sid] = 0.0f;
    return;
  }

  const int m  = lane & 15;
  const int hi = lane >> 4;
  const _Float16* arow = X + ((size_t)r * 16 + m) * DIM;
  const v16h a0 = load_operand(arow, hi, 0);
  const v16h a1 = load_operand(arow, hi, 1);

  float s = 0.0f;
#pragma unroll
  for (int j = 0; j < TILES_PER_STRIP; ++j) {
    const unsigned c = c0 + j;
    const float wgt = (c < r) ? 0.0f : ((c == r) ? 1.0f : 2.0f);
    const _Float16* brow = X + ((size_t)c * 16 + m) * DIM;
    v16h b0 = load_operand(brow, hi, 0);
    v16h b1 = load_operand(brow, hi, 1);

    v8f acc = {};
    acc = __builtin_amdgcn_wmma_f32_16x16x32_f16(false, a0, false, b0,
                                                 (short)0, acc, false, false);
    acc = __builtin_amdgcn_wmma_f32_16x16x32_f16(false, a1, false, b1,
                                                 (short)0, acc, false, false);

    float sj = 0.0f;
#pragma unroll
    for (int k = 0; k < 8; ++k) {
      float g = acc[k];
      float t = fminf(fmaf(g, C4, -C4), 0.0f);   // min((4g-4)*log2e, 0)
      sj += __builtin_amdgcn_exp2f(t);           // raw v_exp_f32 (2^t)
    }
    s = fmaf(wgt, sj, s);
  }

#pragma unroll
  for (int off = 16; off >= 1; off >>= 1) s += __shfl_xor(s, off, 32);
  if (lane == 0) part[sid] = s;
}

// ---------------------------------------------------------------------------
// Kernel 3: single-block deterministic reductions + final scalar.
// ---------------------------------------------------------------------------
__device__ __forceinline__ float block_reduce256(float v, float* red) {
  red[threadIdx.x] = v;
  __syncthreads();
  for (int s = 128; s > 0; s >>= 1) {
    if (threadIdx.x < s) red[threadIdx.x] += red[threadIdx.x + s];
    __syncthreads();
  }
  float r = red[0];
  __syncthreads();
  return r;
}

__global__ void finalize_kernel(const float* __restrict__ partU,
                                const float* __restrict__ partI,
                                const float* __restrict__ alignPart,
                                float* __restrict__ out) {
  __shared__ float red[256];
  const int tid = threadIdx.x;

  float su = 0.f;
  for (int i = tid; i < NSTRIPS; i += 256) su += partU[i];
  float totU = block_reduce256(su, red);

  float si = 0.f;
  for (int i = tid; i < NSTRIPS; i += 256) si += partI[i];
  float totI = block_reduce256(si, red);

  float sa = 0.f;
  for (int i = tid; i < 32; i += 256) sa += alignPart[i];
  float totA = block_reduce256(sa, red);

  if (tid == 0) {
    const float n = (float)N_ROWS;
    const float denom = n * (n - 1.0f);
    float meanU = (totU - n) / denom;
    float meanI = (totI - n) / denom;
    float uniform = 0.5f * (logf(meanU + EPS_U) + logf(meanI + EPS_U));
    out[0] = totA / n + uniform;
  }
}

// ---------------------------------------------------------------------------
extern "C" void kernel_launch(void* const* d_in, const int* in_sizes, int n_in,
                              void* d_out, int out_size, void* d_ws, size_t ws_size,
                              hipStream_t stream) {
  (void)in_sizes; (void)n_in; (void)out_size; (void)ws_size;
  const float* U = (const float*)d_in[0];
  const float* I = (const float*)d_in[1];

  char* ws = (char*)d_ws;
  const size_t szHalf = (size_t)N_ROWS * DIM * sizeof(_Float16);   // 1 MB each
  _Float16* U16 = (_Float16*)(ws);
  _Float16* I16 = (_Float16*)(ws + szHalf);
  float* partU = (float*)(ws + 2 * szHalf);
  float* partI = (float*)(ws + 2 * szHalf + (size_t)NSTRIPS * sizeof(float));
  float* alignPart = (float*)(ws + 2 * szHalf + 2 * (size_t)NSTRIPS * sizeof(float));

  norm_align_kernel<<<N_ROWS / 256, 256, 0, stream>>>(U, I, U16, I16, alignPart);
  gram_exp_kernel<<<dim3(NSTRIPS / WAVES_PER_BLOCK, 2), 256, 0, stream>>>(
      U16, I16, partU, partI);
  finalize_kernel<<<1, 256, 0, stream>>>(partU, partI, alignPart, (float*)d_out);
}
